// AutoEncoder_42898133353152
// MI455X (gfx1250) — matrix operand order repaired and checked by reading, not compiled, not run on hardware
//
#include <hip/hip_runtime.h>

// ---------------------------------------------------------------------------
// PointNet-AE forward for MI455X (gfx1250), wave32 + WMMA bf16 + async-to-LDS.
// Encoder 1x1-convs are GEMMs over B*N=65536 points.
// All conv/fc biases cancel under train-mode BN (constant shift) -> dropped.
// Per layer: GEMM (raw bf16 out + channel sum/sumsq via shuffle+atomics) ->
// finalize (scale,bias) -> in-place normalize (BN+ReLU, bf16) -> next GEMM
// reads normalized data directly, keeping the WMMA loop free of VALU work.
// B tiles are staged to LDS once per block per K-step with
// global_load_async_to_lds_b128 (ASYNCcnt) and consumed via ds_load_b128.
// ---------------------------------------------------------------------------

typedef __attribute__((ext_vector_type(16))) __bf16 v16bf;
typedef __attribute__((ext_vector_type(8)))  __bf16 v8bf;
typedef __attribute__((ext_vector_type(8)))  float  v8f;

__device__ __forceinline__ float bf2f(__bf16 h) {
  unsigned short u = __builtin_bit_cast(unsigned short, h);
  unsigned v = ((unsigned)u) << 16;
  return __builtin_bit_cast(float, v);
}
__device__ __forceinline__ __bf16 f2bf(float f) {
  unsigned u = __builtin_bit_cast(unsigned, f);
  u += 0x7FFFu + ((u >> 16) & 1u);           // round to nearest even
  unsigned short h = (unsigned short)(u >> 16);
  return __builtin_bit_cast(__bf16, h);
}
// LDS byte-offset of a __shared__ pointer (for async-to-LDS VDST operand)
__device__ __forceinline__ unsigned lds_off(const void* p) {
  return (unsigned)(size_t)(__attribute__((address_space(3))) const char*)p;
}

// ----------------------------- small utilities -----------------------------

__global__ __launch_bounds__(256) void zero_f32(float* __restrict__ p, int n) {
  int i = blockIdx.x * 256 + threadIdx.x;
  if (i < n) p[i] = 0.0f;
}

__global__ __launch_bounds__(256) void cvt_bf16(const float* __restrict__ in,
                                                __bf16* __restrict__ out, int n) {
  int i = blockIdx.x * 256 + threadIdx.x;
  if (i < n) out[i] = f2bf(in[i]);
}

// mean/var -> per-channel (scale, bias)
__global__ __launch_bounds__(256) void finalize_bn(const float* __restrict__ sum,
                                                   const float* __restrict__ sq,
                                                   const float* __restrict__ g,
                                                   const float* __restrict__ be,
                                                   float2* __restrict__ sb,
                                                   int C, float invN) {
  int c = blockIdx.x * 256 + threadIdx.x;
  if (c >= C) return;
  float m = sum[c] * invN;
  float v = sq[c] * invN - m * m;
  float s = g[c] * rsqrtf(v + 1e-5f);
  sb[c] = make_float2(s, be[c] - m * s);
}

// in-place BN+ReLU on raw bf16 activations [p][C]; one dword (2 channels)/thr
__global__ __launch_bounds__(256) void norm_act_bf16(unsigned* __restrict__ A,
                                                     const float2* __restrict__ sb,
                                                     int C, int total2) {
  int i = blockIdx.x * 256 + threadIdx.x;
  if (i >= total2) return;
  unsigned u = A[i];
  int c0 = (2 * i) % C;                       // C is even
  float2 s0 = sb[c0], s1 = sb[c0 + 1];
  float v0 = bf2f(__builtin_bit_cast(__bf16, (unsigned short)(u & 0xffffu)));
  float v1 = bf2f(__builtin_bit_cast(__bf16, (unsigned short)(u >> 16)));
  v0 = v0 * s0.x + s0.y; v1 = v1 * s1.x + s1.y;
  v0 = v0 > 0.f ? v0 : 0.f; v1 = v1 > 0.f ? v1 : 0.f;
  unsigned short h0 = __builtin_bit_cast(unsigned short, f2bf(v0));
  unsigned short h1 = __builtin_bit_cast(unsigned short, f2bf(v1));
  A[i] = ((unsigned)h1 << 16) | h0;
}

// --------------------------- conv1 (K=3, scalar) ---------------------------
__global__ __launch_bounds__(256) void conv1_kernel(const float* __restrict__ X,
                                                    const float* __restrict__ W1,
                                                    __bf16* __restrict__ Out,
                                                    float* __restrict__ statSum,
                                                    float* __restrict__ statSq) {
  __shared__ float lsum[64];
  __shared__ float lsq[64];
  int t = threadIdx.x;
  if (t < 64) { lsum[t] = 0.f; lsq[t] = 0.f; }
  __syncthreads();
  int o = t & 63;
  int p = blockIdx.x * 4 + (t >> 6);
  float y = X[p * 3 + 0] * W1[o * 3 + 0] +
            X[p * 3 + 1] * W1[o * 3 + 1] +
            X[p * 3 + 2] * W1[o * 3 + 2];
  Out[(size_t)p * 64 + o] = f2bf(y);
  atomicAdd(&lsum[o], y);
  atomicAdd(&lsq[o], y * y);
  __syncthreads();
  if (t < 64) { atomicAdd(statSum + t, lsum[t]); atomicAdd(statSq + t, lsq[t]); }
}

// ----------------------- WMMA GEMM (inputs pre-normed) ----------------------
// Act:[65536][K] normalized bf16, W:[M][K] bf16 -> Out:[65536][M] raw bf16.
// Block 256 thr = 8 waves; block tile 128(M) x 128(N); wave tile 16x128.
// B tile (128 cols x 32 K = 8KB) staged via global_load_async_to_lds_b128.
__global__ __launch_bounds__(256) void gemm_wmma_bf16(
    const __bf16* __restrict__ Act, const __bf16* __restrict__ W,
    __bf16* __restrict__ Out, float* __restrict__ statSum,
    float* __restrict__ statSq, int M, int K) {
  __shared__ __bf16 tileB[128 * 32];          // [col][k], 8 KB

  const int lane = threadIdx.x & 31;
  const int wave = threadIdx.x >> 5;
  const int hf   = lane >> 4;
  const int l15  = lane & 15;
  const int mBase = blockIdx.x * 128 + wave * 16;
  const bool mAct = (mBase < M);              // wave-uniform
  const int nBase = blockIdx.y * 128;

  // staging: thread i -> col = i>>1, 16-elem K half = (i&1)*16 (32 B)
  const int sCol = threadIdx.x >> 1;
  const int sK   = (threadIdx.x & 1) * 16;
  const __bf16* gsrc = Act + (size_t)(nBase + sCol) * K + sK;
  const unsigned ldsDst = lds_off(&tileB[sCol * 32 + sK]);

  v8f acc[8] = {};

  for (int k0 = 0; k0 < K; k0 += 32) {
    // async stage B tile (ASYNCcnt), then make visible to all waves
    asm volatile("global_load_async_to_lds_b128 %0, %1, off"
                 :: "v"(ldsDst), "v"(gsrc) : "memory");
    asm volatile("global_load_async_to_lds_b128 %0, %1, off offset:16"
                 :: "v"(ldsDst), "v"(gsrc) : "memory");
    asm volatile("s_wait_asynccnt 0x0" ::: "memory");
    __syncthreads();

    if (mAct) {
      // A fragment: row (mBase+l15), one contiguous 16-elem K run
      const __bf16* wrow = W + (size_t)(mBase + l15) * K + k0 + hf * 16;
      v8bf alo = *(const v8bf*)(wrow);
      v8bf ahi = *(const v8bf*)(wrow + 8);
      v16bf a;
#pragma unroll
      for (int i = 0; i < 8; ++i) { a[i] = alo[i]; a[8 + i] = ahi[i]; }
#pragma unroll
      for (int t = 0; t < 8; ++t) {
        // B fragment straight from LDS (ds_load_b128 x2), no VALU work
        v16bf b = *(const v16bf*)(&tileB[(16 * t + l15) * 32 + hf * 16]);
        acc[t] = __builtin_amdgcn_wmma_f32_16x16x32_bf16(
            false, a, false, b, (short)0, acc[t], false, false);
      }
    }
    __syncthreads();                          // protect LDS before next stage
    gsrc += 32;
  }

  if (!mAct) return;

  // epilogue 1: per-channel sum/sumsq (reduce 128 n's per row)
#pragma unroll
  for (int r = 0; r < 8; ++r) {
    float s = 0.f, q = 0.f;
#pragma unroll
    for (int t = 0; t < 8; ++t) { float v = acc[t][r]; s += v; q += v * v; }
#pragma unroll
    for (int msk = 1; msk < 16; msk <<= 1) {
      s += __shfl_xor(s, msk, 32);
      q += __shfl_xor(q, msk, 32);
    }
    if (l15 == 0) {
      atomicAdd(statSum + mBase + r + 8 * hf, s);
      atomicAdd(statSq  + mBase + r + 8 * hf, q);
    }
  }

  // epilogue 2: store raw bf16 output [point][channel], paired rows
#pragma unroll
  for (int t = 0; t < 8; ++t) {
    int n = nBase + 16 * t + l15;
    __bf16* orow = Out + (size_t)n * M + mBase + 8 * hf;
#pragma unroll
    for (int r = 0; r < 8; r += 2) {
      unsigned short u0 = __builtin_bit_cast(unsigned short, f2bf(acc[t][r]));
      unsigned short u1 = __builtin_bit_cast(unsigned short, f2bf(acc[t][r + 1]));
      *(unsigned*)(orow + r) = ((unsigned)u1 << 16) | u0;
    }
  }
}

// ------------------- max-pool over 2048 points (normalized) -----------------
__global__ __launch_bounds__(256) void pool_kernel(const __bf16* __restrict__ Act,
                                                   float* __restrict__ pooled,
                                                   int C, int off) {
  int idx = blockIdx.x * 256 + threadIdx.x;
  if (idx >= C * 32) return;
  int c = idx % C, b = idx / C;
  float mx = 0.f;                             // post-ReLU values are >= 0
  const __bf16* base = Act + (size_t)b * 2048 * C + c;
  for (int n = 0; n < 2048; ++n) {
    float v = bf2f(base[(size_t)n * C]);
    mx = v > mx ? v : mx;
  }
  pooled[b * 1920 + off + c] = mx;
}

// --------------------------- small dense layers ----------------------------
__global__ __launch_bounds__(256) void fc_kernel(const float* __restrict__ In,
                                                 const float* __restrict__ W,
                                                 float* __restrict__ Out,
                                                 int Fin, int Fout) {
  int idx = blockIdx.x * 256 + threadIdx.x;
  if (idx >= 32 * Fout) return;
  int o = idx % Fout, b = idx / Fout;
  const float* ip = In + (size_t)b * Fin;
  const float* wp = W + (size_t)o * Fin;
  float s = 0.f;
  for (int k = 0; k < Fin; ++k) s += ip[k] * wp[k];
  Out[idx] = s;
}

// per-feature stats over batch: element (b,c,l) at b*strideB + c*strideC + l
__global__ __launch_bounds__(256) void stats_kernel(const float* __restrict__ X,
                                                    float* __restrict__ sum,
                                                    float* __restrict__ sq,
                                                    int C, int L, int strideB,
                                                    int strideC) {
  int c = blockIdx.x * 256 + threadIdx.x;
  if (c >= C) return;
  float s = 0.f, q = 0.f;
  for (int b = 0; b < 32; ++b) {
    const float* p = X + (size_t)b * strideB + (size_t)c * strideC;
    for (int l = 0; l < L; ++l) { float v = p[l]; s += v; q += v * v; }
  }
  sum[c] = s; sq[c] = q;
}

__global__ __launch_bounds__(256) void norm_relu(const float* __restrict__ In,
                                                 const float2* __restrict__ sb,
                                                 float* __restrict__ Out,
                                                 int n, int dv, int md) {
  int i = blockIdx.x * 256 + threadIdx.x;
  if (i >= n) return;
  float2 s = sb[(i / dv) % md];
  float v = In[i] * s.x + s.y;
  Out[i] = v > 0.f ? v : 0.f;
}

// decoder 1x1 conv: out[b][o][l] = sum_i In[b][i][l] * W[o][i], i<64, l<128
__global__ __launch_bounds__(256) void conv_small(const float* __restrict__ In,
                                                  const float* __restrict__ W,
                                                  float* __restrict__ Out, int O) {
  int idx = blockIdx.x * 256 + threadIdx.x;
  if (idx >= 32 * O * 128) return;
  int l = idx % 128, o = (idx / 128) % O, b = idx / (128 * O);
  const float* ip = In + (size_t)b * 8192 + l;
  const float* wp = W + o * 64;
  float s = 0.f;
  for (int i = 0; i < 64; ++i) s += ip[i * 128] * wp[i];
  Out[idx] = s;
}

// final BN (no ReLU) -> d_out
__global__ __launch_bounds__(256) void final_bn(const float* __restrict__ In,
                                                const float2* __restrict__ sb,
                                                float* __restrict__ Out, int n) {
  int i = blockIdx.x * 256 + threadIdx.x;
  if (i >= n) return;
  float2 s = sb[(i / 128) % 48];
  Out[i] = In[i] * s.x + s.y;
}

// ---------------------------------------------------------------------------

extern "C" void kernel_launch(void* const* d_in, const int* in_sizes, int n_in,
                              void* d_out, int out_size, void* d_ws, size_t ws_size,
                              hipStream_t stream) {
  (void)in_sizes; (void)n_in; (void)out_size; (void)ws_size;

  // inputs in setup_inputs() flat order: x, labels, then params in dict order
  const float* x    = (const float*)d_in[0];
  const float* cw1  = (const float*)d_in[2];
  const float* g1   = (const float*)d_in[4];
  const float* be1  = (const float*)d_in[5];
  const float* cw[5]; const float* gg[5]; const float* bb[5];
  for (int i = 0; i < 5; ++i) {
    cw[i] = (const float*)d_in[6 + 4 * i];
    gg[i] = (const float*)d_in[8 + 4 * i];
    bb[i] = (const float*)d_in[9 + 4 * i];
  }
  const float* fc1w  = (const float*)d_in[26];
  const float* fg1   = (const float*)d_in[28];
  const float* fbe1  = (const float*)d_in[29];
  const float* fc11w = (const float*)d_in[30];
  const float* g11   = (const float*)d_in[32];
  const float* be11  = (const float*)d_in[33];
  const float* c12w  = (const float*)d_in[34];
  const float* g12   = (const float*)d_in[36];
  const float* be12  = (const float*)d_in[37];
  const float* c13w  = (const float*)d_in[38];
  const float* g13   = (const float*)d_in[40];
  const float* be13  = (const float*)d_in[41];

  const int Ms[5] = {64, 128, 256, 512, 1024};
  const int Ks[5] = {64, 64, 128, 256, 512};
  const int NP = 65536;  // B*N

  // -------- workspace carve-up (~209 MB) --------
  char* ws = (char*)d_ws;
  size_t off = 0;
  auto take = [&](size_t bytes) -> char* {
    char* p = ws + off;
    off = (off + bytes + 255) & ~(size_t)255;
    return p;
  };
  __bf16* bufs[2];
  bufs[0] = (__bf16*)take((size_t)NP * 512 * 2);    // 64/128/512-ch acts
  bufs[1] = (__bf16*)take((size_t)NP * 1024 * 2);   // 64/256/1024-ch acts
  __bf16* wbf[5];
  for (int i = 0; i < 5; ++i) wbf[i] = (__bf16*)take((size_t)Ms[i] * Ks[i] * 2);
  float* statSum = (float*)take(16384 * 4);         // sum[8192] + sq[8192]
  float* statSq  = statSum + 8192;
  float2* sbL[6];
  const int sbC[6] = {64, 64, 128, 256, 512, 1024};
  for (int i = 0; i < 6; ++i) sbL[i] = (float2*)take((size_t)sbC[i] * 8);
  float2* sbF1  = (float2*)take(1024 * 8);
  float2* sbF11 = (float2*)take(8192 * 8);
  float2* sb12  = (float2*)take(64 * 8);
  float2* sb13  = (float2*)take(48 * 8);
  float* pooled = (float*)take(32 * 1920 * 4);
  float* latraw = (float*)take(32 * 1024 * 4);
  float* latN   = (float*)take(32 * 1024 * 4);
  float* pcraw  = (float*)take(32 * 8192 * 4);
  float* pcN    = (float*)take(32 * 8192 * 4);
  float* c12raw = (float*)take(32 * 64 * 128 * 4);
  float* c12N   = (float*)take(32 * 64 * 128 * 4);
  float* c13raw = (float*)take(32 * 48 * 128 * 4);

  // -------- weights fp32 -> bf16 --------
  for (int i = 0; i < 5; ++i) {
    int n = Ms[i] * Ks[i];
    cvt_bf16<<<(n + 255) / 256, 256, 0, stream>>>(cw[i], wbf[i], n);
  }

  // -------- conv1 (K=3) + BN stats + in-place normalize --------
  zero_f32<<<64, 256, 0, stream>>>(statSum, 16384);
  conv1_kernel<<<NP / 4, 256, 0, stream>>>(x, cw1, bufs[0], statSum, statSq);
  finalize_bn<<<1, 256, 0, stream>>>(statSum, statSq, g1, be1, sbL[0], 64,
                                     1.f / NP);
  norm_act_bf16<<<(NP * 64 / 2 + 255) / 256, 256, 0, stream>>>(
      (unsigned*)bufs[0], sbL[0], 64, NP * 64 / 2);

  // -------- encoder WMMA GEMM chain --------
  for (int i = 0; i < 5; ++i) {
    const __bf16* aIn = bufs[i & 1];
    __bf16* aOut = bufs[(i + 1) & 1];
    zero_f32<<<64, 256, 0, stream>>>(statSum, 16384);
    dim3 grid((Ms[i] + 127) / 128, NP / 128);
    gemm_wmma_bf16<<<grid, 256, 0, stream>>>(aIn, wbf[i], aOut, statSum, statSq,
                                             Ms[i], Ks[i]);
    finalize_bn<<<(Ms[i] + 255) / 256, 256, 0, stream>>>(
        statSum, statSq, gg[i], bb[i], sbL[i + 1], Ms[i], 1.f / NP);
    norm_act_bf16<<<(NP * Ms[i] / 2 + 255) / 256, 256, 0, stream>>>(
        (unsigned*)aOut, sbL[i + 1], Ms[i], NP * Ms[i] / 2);
    if (i >= 1) {  // pool x128/x256/x512/x1024 at concat offsets
      int po = (i == 1) ? 1792 : (i == 2) ? 1536 : (i == 3) ? 1024 : 0;
      pool_kernel<<<(Ms[i] * 32 + 255) / 256, 256, 0, stream>>>(
          aOut, pooled, Ms[i], po);
    }
  }

  // -------- fc1: 1920 -> 1024, BN over batch --------
  fc_kernel<<<(32 * 1024 + 255) / 256, 256, 0, stream>>>(pooled, fc1w, latraw,
                                                         1920, 1024);
  stats_kernel<<<4, 256, 0, stream>>>(latraw, statSum, statSq, 1024, 1, 1024, 1);
  finalize_bn<<<4, 256, 0, stream>>>(statSum, statSq, fg1, fbe1, sbF1, 1024,
                                     1.f / 32.f);
  norm_relu<<<(32 * 1024 + 255) / 256, 256, 0, stream>>>(latraw, sbF1, latN,
                                                         32 * 1024, 1, 1024);

  // -------- fc11: 1024 -> 8192, BN over batch --------
  fc_kernel<<<(32 * 8192 + 255) / 256, 256, 0, stream>>>(latN, fc11w, pcraw,
                                                         1024, 8192);
  stats_kernel<<<32, 256, 0, stream>>>(pcraw, statSum, statSq, 8192, 1, 8192, 1);
  finalize_bn<<<32, 256, 0, stream>>>(statSum, statSq, g11, be11, sbF11, 8192,
                                      1.f / 32.f);
  norm_relu<<<(32 * 8192 + 255) / 256, 256, 0, stream>>>(pcraw, sbF11, pcN,
                                                         32 * 8192, 1, 8192);

  // -------- c12: 64 -> 64 over L=128, BN over (b,l) --------
  conv_small<<<(32 * 64 * 128 + 255) / 256, 256, 0, stream>>>(pcN, c12w, c12raw,
                                                              64);
  stats_kernel<<<1, 256, 0, stream>>>(c12raw, statSum, statSq, 64, 128, 8192,
                                      128);
  finalize_bn<<<1, 256, 0, stream>>>(statSum, statSq, g12, be12, sb12, 64,
                                     1.f / 4096.f);
  norm_relu<<<(32 * 64 * 128 + 255) / 256, 256, 0, stream>>>(
      c12raw, sb12, c12N, 32 * 64 * 128, 128, 64);

  // -------- c13: 64 -> 48, BN (no ReLU) -> output --------
  conv_small<<<(32 * 48 * 128 + 255) / 256, 256, 0, stream>>>(c12N, c13w, c13raw,
                                                              48);
  stats_kernel<<<1, 256, 0, stream>>>(c13raw, statSum, statSq, 48, 128, 6144,
                                      128);
  finalize_bn<<<1, 256, 0, stream>>>(statSum, statSq, g13, be13, sb13, 48,
                                     1.f / 4096.f);
  final_bn<<<(32 * 48 * 128 + 255) / 256, 256, 0, stream>>>(
      c13raw, sb13, (float*)d_out, 32 * 48 * 128);
}
